// AuxiliaryConditionerBlock_22222160789917
// MI455X (gfx1250) — matrix operand order repaired and checked by reading, not compiled, not run on hardware
//
#include <hip/hip_runtime.h>

typedef __bf16 bf16_t;
typedef __attribute__((ext_vector_type(16))) __bf16 v16bf;
typedef __attribute__((ext_vector_type(8)))  __bf16 v8bf;
typedef __attribute__((ext_vector_type(8)))  float  v8f;

#define S_LEN    2048
#define NODE_DIM 128
#define NHEAD    24          // 3*H
#define CDIM     32
#define BM       64          // i-rows per block (4 waves x 16)
#define BN       64          // j-tile (4 x 16 logit sub-tiles)
#define TWO_PI   6.283185307179586f

// Build a 16-element bf16 fragment from two 16-byte chunks (lowered to b128 loads).
static __device__ __forceinline__ v16bf frag_from(const bf16_t* lo_p, const bf16_t* hi_p) {
  v8bf lo = *reinterpret_cast<const v8bf*>(lo_p);
  v8bf hi = *reinterpret_cast<const v8bf*>(hi_p);
  return __builtin_shufflevector(lo, hi, 0,1,2,3,4,5,6,7,8,9,10,11,12,13,14,15);
}

// gfx1250 async DMA: global -> LDS, 16B per lane, tracked by ASYNCcnt.
static __device__ __forceinline__ void async_b128(const bf16_t* g, bf16_t* l) {
  unsigned lds_off = (unsigned)(size_t)(void*)l;        // low 32 bits = LDS byte offset
  unsigned long long ga = (unsigned long long)(size_t)(const void*)g;
  asm volatile("global_load_async_to_lds_b128 %0, %1, off"
               :: "v"(lds_off), "v"(ga) : "memory");
}
static __device__ __forceinline__ void wait_async0() {
  asm volatile("s_wait_asynccnt 0" ::: "memory");
}

// ---------------------------------------------------------------- zero output
__global__ void zero_out_kernel(float* __restrict__ out, int n) {
  int i = blockIdx.x * blockDim.x + threadIdx.x;
  if (i < n) out[i] = 0.0f;
}

// --------------------------------------------- value projection, d-major output
__global__ void proj_v_kernel(const float* __restrict__ nodes,
                              const float* __restrict__ w,
                              const float* __restrict__ b,
                              bf16_t* __restrict__ Vt) {
  int idx = blockIdx.x * 256 + threadIdx.x;            // grid sized exactly
  int col = idx >> 11;                                 // / S_LEN
  int s   = idx & (S_LEN - 1);
  const float4* a  = reinterpret_cast<const float4*>(nodes + (size_t)s * NODE_DIM);
  const float4* wr = reinterpret_cast<const float4*>(w + (size_t)col * NODE_DIM);
  float acc = b[col];
#pragma unroll 8
  for (int i = 0; i < NODE_DIM / 4; ++i) {
    float4 x = a[i], y = wr[i];
    acc += x.x * y.x + x.y * y.y + x.z * y.z + x.w * y.w;
  }
  Vt[(size_t)col * S_LEN + s] = (bf16_t)acc;
}

// --------------------------------------------------------- K/Q projection (bf16)
__global__ void proj_kq_kernel(const float* __restrict__ nodes,
                               const float* __restrict__ pos,
                               const float* __restrict__ rot,
                               const float* __restrict__ w_n, const float* __restrict__ b_n,
                               const float* __restrict__ w_p, const float* __restrict__ b_p,
                               const float* __restrict__ w_r,
                               bf16_t* __restrict__ Kg, bf16_t* __restrict__ Qg) {
  int idx = blockIdx.x * 256 + threadIdx.x;            // grid sized exactly
  int s = idx / (NHEAD * CDIM);
  int r = idx % (NHEAD * CDIM);
  int h = r / CDIM;
  int c = r % CDIM;

  float kacc, qacc;
  if (h < 8) {
    int rk = h * 64 + c;
    const float4* a  = reinterpret_cast<const float4*>(nodes + (size_t)s * NODE_DIM);
    const float4* wk = reinterpret_cast<const float4*>(w_n + (size_t)rk * NODE_DIM);
    const float4* wq = reinterpret_cast<const float4*>(w_n + (size_t)(rk + 32) * NODE_DIM);
    kacc = b_n[rk];
    qacc = b_n[rk + 32];
#pragma unroll 8
    for (int i = 0; i < NODE_DIM / 4; ++i) {
      float4 x = a[i], yk = wk[i], yq = wq[i];
      kacc += x.x * yk.x + x.y * yk.y + x.z * yk.z + x.w * yk.w;
      qacc += x.x * yq.x + x.y * yq.y + x.z * yq.z + x.w * yq.w;
    }
  } else if (h < 16) {
    float pf[6];
#pragma unroll
    for (int i = 0; i < 3; ++i) {
      float p = pos[(size_t)s * 3 + i];
      pf[i]     = cosf(TWO_PI * p);
      pf[i + 3] = sinf(TWO_PI * p);
    }
    int rk = (h - 8) * 64 + c;
    kacc = b_p[rk];
    qacc = b_p[rk + 32];
#pragma unroll
    for (int i = 0; i < 6; ++i) {
      kacc += pf[i] * w_p[(size_t)rk * 6 + i];
      qacc += pf[i] * w_p[(size_t)(rk + 32) * 6 + i];
    }
  } else {
    int rk = (h - 16) * 64 + c;
    kacc = 0.0f;
    qacc = 0.0f;
#pragma unroll
    for (int i = 0; i < 4; ++i) {
      float rv = rot[(size_t)s * 4 + i];
      kacc += rv * w_r[(size_t)rk * 4 + i];
      qacc += rv * w_r[(size_t)(rk + 32) * 4 + i];
    }
  }
  size_t o = (size_t)s * (NHEAD * CDIM) + h * CDIM + c;
  Kg[o] = (bf16_t)kacc;
  Qg[o] = (bf16_t)qacc;
}

// ------------------------------------------------------------ flash attention
// block = (head h, 64-row i-tile); 4 waves of 32, each wave owns 16 i-rows.
__launch_bounds__(128)
__global__ void attn_kernel(const bf16_t* __restrict__ Kg,
                            const bf16_t* __restrict__ Qg,
                            const bf16_t* __restrict__ Vt,   // [h*128+d][s]
                            float* __restrict__ out) {
  __shared__ __align__(16) bf16_t Qs[BN * CDIM];        // [j][c]    4 KB
  __shared__ __align__(16) bf16_t VsT[NODE_DIM * BN];   // [d][j]   16 KB
  __shared__ __align__(16) bf16_t Ps[4 * 16 * BN];      // per-wave P  8 KB

  const int tid  = threadIdx.x;
  const int wave = tid >> 5;
  const int lane = tid & 31;
  const int half = lane >> 4;      // 0: lanes 0-15, 1: lanes 16-31
  const int lmod = lane & 15;
  const int h    = blockIdx.y;
  const int i0   = blockIdx.x * BM + wave * 16;
  const bool rot_head = (h >= 16);

  // --- K fragment: WMMA A-matrix 16x32 bf16.
  const bf16_t* kr = Kg + (size_t)(i0 + lmod) * (NHEAD * CDIM) + h * CDIM + half * 8;
  v16bf kf = frag_from(kr, kr + 16);

  v8f acc[8];
#pragma unroll
  for (int t = 0; t < 8; ++t) acc[t] = (v8f){};
  float rowmax[8], rowsum[8];
#pragma unroll
  for (int r = 0; r < 8; ++r) { rowmax[r] = -3.0e38f; rowsum[r] = 0.0f; }

  bf16_t* myPs = Ps + wave * 16 * BN;

  for (int j0 = 0; j0 < S_LEN; j0 += BN) {
    // ---- async DMA staging of Q (64x32) and V (128x64, d-major) tiles into LDS
    {
      // Q: 256 16B-chunks, 2 per thread
#pragma unroll
      for (int k = 0; k < 2; ++k) {
        int id = tid + 128 * k;
        int rowj = id >> 2, chunk = id & 3;
        async_b128(Qg + (size_t)(j0 + rowj) * (NHEAD * CDIM) + h * CDIM + chunk * 8,
                   Qs + rowj * CDIM + chunk * 8);
      }
      // V: thread d copies 64 j-halfs (128B) as 8 async b128
      const bf16_t* vsrc = Vt + ((size_t)h * NODE_DIM + tid) * S_LEN + j0;
      bf16_t* vdst = VsT + tid * BN;
#pragma unroll
      for (int c = 0; c < 8; ++c) async_b128(vsrc + c * 8, vdst + c * 8);
    }
    wait_async0();
    __syncthreads();

    // ---- logits: 4 sub-tiles of 16 j-columns each
    v8f sS[4];
#pragma unroll
    for (int sub = 0; sub < 4; ++sub) {
      const bf16_t* qp = Qs + (sub * 16 + lmod) * CDIM + half * 8;
      v16bf qb = frag_from(qp, qp + 16);
      v8f z = (v8f){};
      sS[sub] = __builtin_amdgcn_wmma_f32_16x16x32_bf16(false, kf, false, qb, (short)0, z, false, false);
    }

    if (rot_head) {
#pragma unroll
      for (int sub = 0; sub < 4; ++sub)
#pragma unroll
        for (int r = 0; r < 8; ++r) sS[sub][r] *= sS[sub][r];
    }

    // ---- online softmax (rows = M = r + 8*half; cols = N = lane within 16)
    float p[4][8];
#pragma unroll
    for (int r = 0; r < 8; ++r) {
      float v = fmaxf(fmaxf(sS[0][r], sS[1][r]), fmaxf(sS[2][r], sS[3][r]));
      v = fmaxf(v, __shfl_xor(v, 1, 32));
      v = fmaxf(v, __shfl_xor(v, 2, 32));
      v = fmaxf(v, __shfl_xor(v, 4, 32));
      v = fmaxf(v, __shfl_xor(v, 8, 32));
      float mnew  = fmaxf(rowmax[r], v);
      float alpha = __expf(rowmax[r] - mnew);
      rowmax[r] = mnew;
      float ps = 0.0f;
#pragma unroll
      for (int sub = 0; sub < 4; ++sub) {
        p[sub][r] = __expf(sS[sub][r] - mnew);
        ps += p[sub][r];
      }
      ps += __shfl_xor(ps, 1, 32);
      ps += __shfl_xor(ps, 2, 32);
      ps += __shfl_xor(ps, 4, 32);
      ps += __shfl_xor(ps, 8, 32);
      rowsum[r] = rowsum[r] * alpha + ps;
#pragma unroll
      for (int t = 0; t < 8; ++t) acc[t][r] *= alpha;
    }

    // ---- transpose P (C/D layout -> A layout) through per-wave LDS
#pragma unroll
    for (int r = 0; r < 8; ++r) {
      int rowbase = (r + half * 8) * BN;
#pragma unroll
      for (int sub = 0; sub < 4; ++sub)
        myPs[rowbase + sub * 16 + lmod] = (bf16_t)p[sub][r];
    }
    __syncthreads();

    // A fragments of P: K=j 0..31 and 32..63
    const bf16_t* pp = myPs + lmod * BN + half * 8;
    v16bf pa0 = frag_from(pp,      pp + 16);
    v16bf pa1 = frag_from(pp + 32, pp + 48);

    // ---- O(16x128) += P(16x64) @ V(64x128): 16 WMMAs over d-tiles
#pragma unroll
    for (int t = 0; t < 8; ++t) {
      const bf16_t* vp = VsT + (t * 16 + lmod) * BN + half * 8;
      v16bf vb0 = frag_from(vp,      vp + 16);
      v16bf vb1 = frag_from(vp + 32, vp + 48);
      acc[t] = __builtin_amdgcn_wmma_f32_16x16x32_bf16(false, pa0, false, vb0, (short)0, acc[t], false, false);
      acc[t] = __builtin_amdgcn_wmma_f32_16x16x32_bf16(false, pa1, false, vb1, (short)0, acc[t], false, false);
    }
    __syncthreads();
  }

  // ---- normalize and accumulate across heads
#pragma unroll
  for (int r = 0; r < 8; ++r) {
    float inv = 1.0f / rowsum[r];
    int row = i0 + r + half * 8;
#pragma unroll
    for (int t = 0; t < 8; ++t) {
      atomicAdd(&out[(size_t)row * NODE_DIM + t * 16 + lmod], acc[t][r] * inv);
    }
  }
}

// ------------------------------------------------------------------- launcher
extern "C" void kernel_launch(void* const* d_in, const int* in_sizes, int n_in,
                              void* d_out, int out_size, void* d_ws, size_t ws_size,
                              hipStream_t stream) {
  (void)in_sizes; (void)n_in; (void)out_size; (void)ws_size;
  const float* nodes      = (const float*)d_in[0];
  const float* pos        = (const float*)d_in[1];
  const float* rot        = (const float*)d_in[2];
  const float* w_nodes_kq = (const float*)d_in[3];
  const float* b_nodes_kq = (const float*)d_in[4];
  const float* w_pos_kq   = (const float*)d_in[5];
  const float* b_pos_kq   = (const float*)d_in[6];
  const float* w_rot_kq   = (const float*)d_in[7];
  const float* w_values   = (const float*)d_in[8];
  const float* b_values   = (const float*)d_in[9];
  float* out = (float*)d_out;

  char* ws = (char*)d_ws;
  bf16_t* Kg = (bf16_t*)(ws);
  bf16_t* Qg = (bf16_t*)(ws + (size_t)S_LEN * NHEAD * CDIM * 2);
  bf16_t* Vt = (bf16_t*)(ws + (size_t)S_LEN * NHEAD * CDIM * 4);

  zero_out_kernel<<<(S_LEN * NODE_DIM + 255) / 256, 256, 0, stream>>>(out, S_LEN * NODE_DIM);
  proj_kq_kernel<<<(S_LEN * NHEAD * CDIM) / 256, 256, 0, stream>>>(
      nodes, pos, rot, w_nodes_kq, b_nodes_kq, w_pos_kq, b_pos_kq, w_rot_kq, Kg, Qg);
  proj_v_kernel<<<(S_LEN * NHEAD * NODE_DIM) / 256, 256, 0, stream>>>(
      nodes, w_values, b_values, Vt);

  dim3 grid(S_LEN / BM, NHEAD);
  attn_kernel<<<grid, 128, 0, stream>>>(Kg, Qg, Vt, out);
}
